// TokenAlignerOT_6322191859876
// MI455X (gfx1250) — matrix operand order repaired and checked by reading, not compiled, not run on hardware
//
#include <hip/hip_runtime.h>
#include <hip/hip_bf16.h>
#include <cstdint>

// ---------------------------------------------------------------------------
// TokenAlignerOT on gfx1250 (MI455X):
//   A = cos(X,Y)  via f16 WMMA GEMM (f32 accum), async-LDS double-buffered
//   K = exp(10A) stored f16 (+ explicit f16 K^T) -> A+K+KT = 128MB, L2-resident
//   250 Sinkhorn iters: wave-per-row matvecs (f16 reads, f32 accum) from L2
//   T = A*(u*K*v)+delta in f16 (reuses KT buffer), out = T @ X via WMMA GEMM
// ---------------------------------------------------------------------------

typedef _Float16 h8  __attribute__((ext_vector_type(8)));
typedef _Float16 h16 __attribute__((ext_vector_type(16)));
typedef float    f8  __attribute__((ext_vector_type(8)));

#define N_TOK   4096
#define D_EMB   2048
#define N_ITER  250
#define FI      (0.001f / (0.001f + 0.1f))   // reg_m / (reg_m + reg)
#define INV_REG 10.0f                        // 1 / reg
#define MARG    (1.0f / 4096.0f)             // a_i = b_j = 1/n
#define LDH     40                           // LDS row pitch in halves (32 + 8 pad)

// ---------------------------- helpers --------------------------------------

__device__ __forceinline__ float wave_reduce_sum(float s) {
#pragma unroll
  for (int off = 16; off > 0; off >>= 1) s += __shfl_xor(s, off, 32);
  return s;
}

// Async copy 16B from global to LDS (memory -> LDS, no VGPR roundtrip).
// VDST VGPR carries the LDS byte address (low 32 bits of the flat address).
__device__ __forceinline__ void async_copy_b128(const void* gsrc, void* ldst) {
  uint32_t lds_off = (uint32_t)(uintptr_t)ldst;
  asm volatile("global_load_async_to_lds_b128 %0, %1, off"
               :: "v"(lds_off), "v"(gsrc) : "memory");
}

__device__ __forceinline__ void wait_asynccnt_le4() {
  asm volatile("s_wait_asynccnt 0x4" ::: "memory");
}
__device__ __forceinline__ void wait_asynccnt_0() {
  asm volatile("s_wait_asynccnt 0x0" ::: "memory");
}

// A fragment (16x32 f16, MxK). Per ISA 7.12.2:
//   lane L -> M = L%16 ; halves 0..7  -> K = ((L>=16)?8:0) + h
//                        halves 8..15 -> K = 16 + ((L>=16)?8:0) + (h-8)
// => two contiguous 16B LDS reads per lane.
__device__ __forceinline__ h16 load_fragA(const _Float16* tile, int lane) {
  int m  = lane & 15;
  int k0 = (lane >> 4) << 3;
  const _Float16* p = tile + m * LDH;
  h8 lo = *(const h8*)(p + k0);
  h8 hi = *(const h8*)(p + 16 + k0);
  h16 r;
#pragma unroll
  for (int i = 0; i < 8; ++i) { r[i] = lo[i]; r[i + 8] = hi[i]; }
  return r;
}

// B fragment (32x16 f16, KxN), staged in LDS as [n][k]:
//   lane L -> N = L%16 ; halves 0..7 -> K = ((L>=16)?8:0) + h, halves 8..15 -> K = 16 + ((L>=16)?8:0) + (h-8)
__device__ __forceinline__ h16 load_fragB(const _Float16* tile, int lane) {
  int n  = lane & 15;
  int k0 = (lane >> 4) << 3;
  const _Float16* p = tile + n * LDH + k0;
  h8 lo = *(const h8*)(p);
  h8 hi = *(const h8*)(p + 16);
  h16 r;
#pragma unroll
  for (int i = 0; i < 8; ++i) { r[i] = lo[i]; r[i + 8] = hi[i]; }
  return r;
}

// ------------------------- small prep kernels ------------------------------

// One wave per row: L2-normalize to f16.
__global__ __launch_bounds__(256)
void rownorm_kernel(const float* __restrict__ src, _Float16* __restrict__ dstN,
                    int rows, int cols) {
  int wave = (blockIdx.x * blockDim.x + threadIdx.x) >> 5;
  int lane = threadIdx.x & 31;
  if (wave >= rows) return;
  const float* r = src + (size_t)wave * cols;
  float ss = 0.f;
  for (int j = lane; j < cols; j += 32) { float x = r[j]; ss += x * x; }
  ss = wave_reduce_sum(ss);
  float inv = 1.0f / fmaxf(sqrtf(ss), 1e-8f);
  _Float16* dn = dstN + (size_t)wave * cols;
  for (int j = lane; j < cols; j += 32) dn[j] = (_Float16)(r[j] * inv);
}

// f32 [rows][cols] -> f16 [cols][rows] (64x64 LDS tiles, coalesced both ways)
__global__ __launch_bounds__(256)
void transpose_cvt_kernel(const float* __restrict__ src, _Float16* __restrict__ dstT,
                          int rows, int cols) {
  __shared__ _Float16 t[64][65];
  int bx = blockIdx.x * 64;  // col base
  int by = blockIdx.y * 64;  // row base
#pragma unroll
  for (int l = 0; l < 16; ++l) {
    int lin = threadIdx.x + 256 * l;
    int rr = lin >> 6, cc = lin & 63;
    t[cc][rr] = (_Float16)src[(size_t)(by + rr) * cols + (bx + cc)];
  }
  __syncthreads();
#pragma unroll
  for (int l = 0; l < 16; ++l) {
    int lin = threadIdx.x + 256 * l;
    int rr = lin >> 6, cc = lin & 63;
    dstT[(size_t)(bx + rr) * rows + (by + cc)] = t[rr][cc];
  }
}

__global__ void init_uv_kernel(float* __restrict__ u, float* __restrict__ v, int n) {
  int i = blockIdx.x * blockDim.x + threadIdx.x;
  if (i < n) { u[i] = 1.0f; v[i] = 1.0f; }
}

// K = exp(10*A) as f16, plus transposed copy KT (64x64 LDS tiles, coalesced).
__global__ __launch_bounds__(256)
void expk_kernel(const float* __restrict__ A, _Float16* __restrict__ K,
                 _Float16* __restrict__ KT, int n) {
  __shared__ _Float16 t[64][65];
  int bx = blockIdx.x * 64;  // col base
  int by = blockIdx.y * 64;  // row base
#pragma unroll
  for (int l = 0; l < 16; ++l) {
    int lin = threadIdx.x + 256 * l;
    int rr = lin >> 6, cc = lin & 63;
    float a = A[(size_t)(by + rr) * n + (bx + cc)];
    _Float16 eh = (_Float16)__expf(INV_REG * a);
    K[(size_t)(by + rr) * n + (bx + cc)] = eh;
    t[cc][rr] = eh;
  }
  __syncthreads();
#pragma unroll
  for (int l = 0; l < 16; ++l) {
    int lin = threadIdx.x + 256 * l;
    int rr = lin >> 6, cc = lin & 63;
    KT[(size_t)(bx + rr) * n + (by + cc)] = t[rr][cc];
  }
}

// ------------------------- Sinkhorn matvec step ----------------------------
// One wave per row of Km (f16, row-major):  s = sum_j Km[row][j]*vin[j]
// then out[row] = (marg / s) ^ fi.  Km rows stream from L2 (K+KT = 64MB).
__global__ __launch_bounds__(256)
void sinkhorn_step_kernel(const _Float16* __restrict__ Km, const float* __restrict__ vin,
                          float* __restrict__ uout, int n) {
  int wave = (blockIdx.x * blockDim.x + threadIdx.x) >> 5;
  int lane = threadIdx.x & 31;
  if (wave >= n) return;
  const h8* row = (const h8*)(Km + (size_t)wave * n);
  float s = 0.f;
  for (int it = lane; it < (n >> 3); it += 32) {
    h8 kv = row[it];
    const float* vv = vin + (it << 3);
#pragma unroll
    for (int h = 0; h < 8; ++h) s += (float)kv[h] * vv[h];
  }
  s = wave_reduce_sum(s);
  if (lane == 0) {
    float r = MARG / fmaxf(s, 1e-30f);
    uout[wave] = __powf(r, FI);
  }
}

// ------------------------- build T (f16) -----------------------------------
// T[i][j] = A[i][j] * (u[i]*K[i][j]*v[j]) + delta[i][j]
__global__ __launch_bounds__(256)
void buildT_kernel(const float* __restrict__ A, const _Float16* __restrict__ K,
                   const float* __restrict__ u, const float* __restrict__ v,
                   const float* __restrict__ delta, _Float16* __restrict__ T, int n) {
  size_t idx  = (size_t)blockIdx.x * blockDim.x + threadIdx.x;
  size_t base = idx << 3;           // 8 elements per thread, same row
  int i = (int)(base >> 12);        // n == 4096
  int j = (int)(base & (size_t)(n - 1));
  float ui = u[i];
#pragma unroll
  for (int h = 0; h < 8; ++h) {
    size_t e = base + h;
    float w = ui * (float)K[e] * v[j + h];
    T[e] = (_Float16)(A[e] * w + delta[e]);
  }
}

// ------------------------- WMMA GEMM ---------------------------------------
// D(f32) = A(f16)[M][K] * B(f16)[N][K]^T.  Block tile 128x128, 8 waves (2Mx4N),
// each wave 4x2 grid of 16x16 tiles, k-step 32, v_wmma_f32_16x16x32_f16.
// Tile staging: GLOBAL_LOAD_ASYNC_TO_LDS_B128 with double buffering (ASYNCcnt).
__global__ __launch_bounds__(256)
void gemm_f16_wmma(const _Float16* __restrict__ Ag, const _Float16* __restrict__ Bg,
                   float* __restrict__ Dg, int Kd, int lda, int ldb, int ldd) {
  __shared__ _Float16 sA[2][128 * LDH];
  __shared__ _Float16 sB[2][128 * LDH];

  const int tid  = threadIdx.x;
  const int lane = tid & 31;
  const int wave = tid >> 5;
  const int wm = wave >> 2;              // 0..1 : 64-row slab
  const int wn = wave & 3;               // 0..3 : 32-col slab
  const int mBase = blockIdx.y * 128;
  const int nBase = blockIdx.x * 128;

  // per-thread staging coordinates: 512 h8-chunks per 128x32 tile, 2 per thread
  int r0 = (tid + 0)   >> 2, kc0 = ((tid + 0)   & 3) << 3;
  int r1 = (tid + 256) >> 2, kc1 = ((tid + 256) & 3) << 3;

  f8 acc[4][2] = {};

  const int nk = Kd >> 5;   // number of 32-wide k tiles

  // prologue: stage tile 0 into buffer 0 (4 async ops per wave)
  async_copy_b128(Ag + (size_t)(mBase + r0) * lda + kc0, sA[0] + r0 * LDH + kc0);
  async_copy_b128(Ag + (size_t)(mBase + r1) * lda + kc1, sA[0] + r1 * LDH + kc1);
  async_copy_b128(Bg + (size_t)(nBase + r0) * ldb + kc0, sB[0] + r0 * LDH + kc0);
  async_copy_b128(Bg + (size_t)(nBase + r1) * ldb + kc1, sB[0] + r1 * LDH + kc1);

  for (int i = 0; i < nk; ++i) {
    const int cur = i & 1;
    if (i + 1 < nk) {
      // prefetch next k-tile into the other buffer (safe: consumed last iter, before its trailing barrier)
      const int k1 = (i + 1) << 5;
      async_copy_b128(Ag + (size_t)(mBase + r0) * lda + k1 + kc0, sA[1 - cur] + r0 * LDH + kc0);
      async_copy_b128(Ag + (size_t)(mBase + r1) * lda + k1 + kc1, sA[1 - cur] + r1 * LDH + kc1);
      async_copy_b128(Bg + (size_t)(nBase + r0) * ldb + k1 + kc0, sB[1 - cur] + r0 * LDH + kc0);
      async_copy_b128(Bg + (size_t)(nBase + r1) * ldb + k1 + kc1, sB[1 - cur] + r1 * LDH + kc1);
      wait_asynccnt_le4();   // async ops retire in order: tile i is resident
    } else {
      wait_asynccnt_0();
    }
    __syncthreads();

    h16 af[4], bf[2];
#pragma unroll
    for (int mt = 0; mt < 4; ++mt)
      af[mt] = load_fragA(sA[cur] + (wm * 64 + mt * 16) * LDH, lane);
#pragma unroll
    for (int nt = 0; nt < 2; ++nt)
      bf[nt] = load_fragB(sB[cur] + (wn * 32 + nt * 16) * LDH, lane);

#pragma unroll
    for (int mt = 0; mt < 4; ++mt)
#pragma unroll
      for (int nt = 0; nt < 2; ++nt)
        acc[mt][nt] = __builtin_amdgcn_wmma_f32_16x16x32_f16(
            false, af[mt], false, bf[nt], (short)0, acc[mt][nt], false, false);

    __syncthreads();
  }

  // store D per ISA C/D layout: VGPR r -> M = r + 8*(lane>=16); N = lane%16
  const int mOff = (lane >> 4) << 3;
  const int nCol = lane & 15;
#pragma unroll
  for (int mt = 0; mt < 4; ++mt)
#pragma unroll
    for (int nt = 0; nt < 2; ++nt) {
      int col = nBase + wn * 32 + nt * 16 + nCol;
#pragma unroll
      for (int r = 0; r < 8; ++r) {
        int row = mBase + wm * 64 + mt * 16 + mOff + r;
        Dg[(size_t)row * ldd + col] = acc[mt][nt][r];
      }
    }
}

// ------------------------- launch ------------------------------------------

extern "C" void kernel_launch(void* const* d_in, const int* in_sizes, int n_in,
                              void* d_out, int out_size, void* d_ws, size_t ws_size,
                              hipStream_t stream) {
  (void)in_sizes; (void)n_in; (void)out_size; (void)ws_size;
  const float* X     = (const float*)d_in[0];   // [4096, 2048]
  const float* Y     = (const float*)d_in[1];   // [4096, 2048]
  const float* delta = (const float*)d_in[2];   // [4096, 4096]
  float* out = (float*)d_out;                   // [4096, 2048] f32

  char* ws = (char*)d_ws;
  const size_t SZ_H_XD = (size_t)N_TOK * D_EMB * sizeof(_Float16);  // 16 MB
  const size_t SZ_F_NN = (size_t)N_TOK * N_TOK * sizeof(float);     // 64 MB
  const size_t SZ_H_NN = (size_t)N_TOK * N_TOK * sizeof(_Float16);  // 32 MB

  _Float16* Xn  = (_Float16*)(ws);                                   // normalized X (f16)
  _Float16* Yn  = (_Float16*)(ws + SZ_H_XD);                         // normalized Y (f16)
  _Float16* XrT = (_Float16*)(ws + 2 * SZ_H_XD);                     // X^T raw (f16 [D][N])
  float*    Am  = (float*)   (ws + 3 * SZ_H_XD);                     // A (f32, 64MB)
  _Float16* Km  = (_Float16*)(ws + 3 * SZ_H_XD + SZ_F_NN);           // K (f16, 32MB)
  _Float16* KT  = (_Float16*)(ws + 3 * SZ_H_XD + SZ_F_NN + SZ_H_NN); // K^T (f16, 32MB)
  float*    u   = (float*)   (ws + 3 * SZ_H_XD + SZ_F_NN + 2 * SZ_H_NN);
  float*    v   = u + N_TOK;
  _Float16* Tm  = KT;  // T reuses K^T storage (dead after Sinkhorn)

  // 1. normalize + f16 conversion; raw-X transpose for GEMM2 B operand
  rownorm_kernel<<<512, 256, 0, stream>>>(X, Xn, N_TOK, D_EMB);
  rownorm_kernel<<<512, 256, 0, stream>>>(Y, Yn, N_TOK, D_EMB);
  transpose_cvt_kernel<<<dim3(D_EMB / 64, N_TOK / 64), 256, 0, stream>>>(X, XrT, N_TOK, D_EMB);
  init_uv_kernel<<<(N_TOK + 255) / 256, 256, 0, stream>>>(u, v, N_TOK);

  // 2. A = Xn @ Yn^T   (B given as [N][K] = Yn)
  gemm_f16_wmma<<<dim3(N_TOK / 128, N_TOK / 128), 256, 0, stream>>>(
      Xn, Yn, Am, D_EMB, D_EMB, D_EMB, N_TOK);

  // 3. K = exp(10A) f16 + explicit transpose
  expk_kernel<<<dim3(N_TOK / 64, N_TOK / 64), 256, 0, stream>>>(Am, Km, KT, N_TOK);

  // 4. Sinkhorn: 250 x (u-step, v-step); K and K^T stay resident in 192MB L2
  for (int it = 0; it < N_ITER; ++it) {
    sinkhorn_step_kernel<<<N_TOK / 8, 256, 0, stream>>>(Km, v, u, N_TOK);
    sinkhorn_step_kernel<<<N_TOK / 8, 256, 0, stream>>>(KT, u, v, N_TOK);
  }

  // 5. T = A*(u*K*v) + delta  (f16, overwrites K^T buffer)
  buildT_kernel<<<(N_TOK * (N_TOK / 8)) / 256, 256, 0, stream>>>(
      Am, Km, u, v, delta, Tm, N_TOK);

  // 6. out = T @ X   (B given as [N][K] = XrT, N = D_EMB)
  gemm_f16_wmma<<<dim3(D_EMB / 128, N_TOK / 128), 256, 0, stream>>>(
      Tm, XrT, out, N_TOK, N_TOK, N_TOK, D_EMB);
}
